// GDN_35364760715593
// MI455X (gfx1250) — compile-verified, matrix operand order checked by reading, not run on hardware
//
#include <hip/hip_runtime.h>
#include <hip/hip_bf16.h>
#include <math.h>

typedef __attribute__((ext_vector_type(16))) _Float16 v16h;
typedef __attribute__((ext_vector_type(8)))  _Float16 v8h;
typedef __attribute__((ext_vector_type(8)))  float    v8f;

#define TOPKK 20
#define EPSB  1e-5f

// ===================== generic WMMA GEMM (f32 in/out, f16 compute) ==========
// C[M,N] = A[M,K] @ B[K,N] (+ bias[N] if bias != nullptr)
// Requires M % 16 == 0 and K % 32 == 0; N arbitrary (masked).
// Block = 128 threads = 4 waves covering 16 rows x 64 cols of C; wave w owns
// the 16x16 tile at cols blockIdx.x*64 + w*16.
// B is staged through LDS: cooperative coalesced f32 loads, f16 convert,
// stored transposed as ldsB[n_local][k] with 80B row stride (16B-aligned rows,
// bank-conflict-free ds_load_b128 fragment reads).
__global__ __launch_bounds__(128) void wmma_gemm_f16(
    const float* __restrict__ A, const float* __restrict__ B,
    const float* __restrict__ bias, float* __restrict__ C,
    int M, int N, int K)
{
  __shared__ __align__(16) _Float16 ldsB[64][40];   // [n_local][k0..31], pad->80B
  const int tid    = threadIdx.x;
  const int lane   = tid & 31;
  const int wave   = tid >> 5;
  const int mbase  = blockIdx.y * 16;
  const int nblock = blockIdx.x * 64;
  const int half16 = (lane >> 4) & 1;   // ISA 7.12.2: lanes16-31 hold K+8 group
  const int l15    = lane & 15;

  union { v8f v; float f[8]; } acc;
  #pragma unroll
  for (int r = 0; r < 8; ++r) acc.f[r] = 0.f;

  // A: row m = l15 of this M-tile; base already offset by half16*8 in K.
  const float* arow = A + (size_t)(mbase + l15) * K + half16 * 8;
  // B fragment source: this lane's column in LDS.
  const v8h* brow = (const v8h*)&ldsB[wave * 16 + l15][0];

  for (int k0 = 0; k0 < K; k0 += 32) {
    // ---- cooperative B panel load: B[k0..k0+32) x [nblock..nblock+64) ----
    #pragma unroll
    for (int it = 0; it < 16; ++it) {
      const int u = it * 128 + tid;
      const int k = u >> 6;            // 0..31
      const int n = u & 63;            // 0..63
      const int col = nblock + n;
      const int csafe = (col < N) ? col : (N - 1);     // clamp addr, no branch
      const float v = B[(size_t)(k0 + k) * N + csafe];
      ldsB[n][k] = (_Float16)((col < N) ? v : 0.f);    // cndmask, not exec mask
    }
    __syncthreads();

    // ---- A fragment: 16 f32 -> f16, K = half16*8 + {0..7, 16..23} ----
    union { v16h v; _Float16 h[16]; } af;
    #pragma unroll
    for (int e = 0; e < 16; ++e) {
      const int kk = (e < 8) ? e : (e + 8);
      af.h[e] = (_Float16)arow[k0 + kk];
    }
    // prefetch next A K-slice (speculative; dropped if past end)
    __builtin_prefetch(arow + k0 + 32, 0, 1);

    // ---- B fragment: two ds_load_b128 (8 halves each), same K striping ----
    const v8h blo = brow[half16];        // halves [half16*8 .. +8)
    const v8h bhi = brow[2 + half16];    // halves [16 + half16*8 .. +8)
    const v16h bv = __builtin_shufflevector(blo, bhi,
        0, 1, 2, 3, 4, 5, 6, 7, 8, 9, 10, 11, 12, 13, 14, 15);

    acc.v = __builtin_amdgcn_wmma_f32_16x16x32_f16(
        false, af.v, false, bv, (short)0, acc.v, false, false);
    __syncthreads();                     // protect LDS before next overwrite
  }

  const int col = nblock + wave * 16 + l15;
  if (col < N) {
    const float bvs = bias ? bias[col] : 0.f;
    #pragma unroll
    for (int r = 0; r < 8; ++r) {
      const int row = mbase + r + half16 * 8;   // C layout: lanes16-31 -> M+8
      C[(size_t)row * N + col] = acc.f[r] + bvs;
    }
  }
}

// ===================== emb row norms =======================================
__global__ void row_norm_kernel(const float* __restrict__ emb,
                                float* __restrict__ nrm, int N, int D)
{
  int i = blockIdx.x * blockDim.x + threadIdx.x;
  if (i >= N) return;
  float s = 0.f;
  for (int d = 0; d < D; ++d) { float v = emb[(size_t)i * D + d]; s += v * v; }
  nrm[i] = sqrtf(s);
}

// ===================== per-row cosine top-20 ===============================
// One block (256 threads) per row i. cos row staged in LDS, then 20 argmax
// passes with low-index tie-break (matches lax.top_k ordering).
__global__ __launch_bounds__(256) void topk_cos_kernel(
    const float* __restrict__ emb, const float* __restrict__ nrm,
    int* __restrict__ topk, int N, int D)
{
  __shared__ float cosv[2048];
  __shared__ float se[64];
  __shared__ float rv[256];
  __shared__ int   ri[256];
  const int i = blockIdx.x, tid = threadIdx.x;
  for (int d = tid; d < D; d += 256) se[d] = emb[(size_t)i * D + d];
  __syncthreads();
  const float ni = nrm[i];
  for (int j = tid; j < 2048; j += 256) {
    float c = -3.0e38f;
    if (j < N) {
      float s = 0.f;
      const float* ej = emb + (size_t)j * D;
      #pragma unroll 8
      for (int d = 0; d < 64; ++d) s += se[d] * ej[d];
      c = s / (ni * nrm[j]);
    }
    cosv[j] = c;
  }
  __syncthreads();
  for (int k = 0; k < TOPKK; ++k) {
    float bv = -3.0e38f; int bi = 0x7fffffff;
    for (int j = tid; j < 2048; j += 256) {
      float c = cosv[j];
      if (c > bv || (c == bv && j < bi)) { bv = c; bi = j; }
    }
    rv[tid] = bv; ri[tid] = bi;
    __syncthreads();
    for (int s = 128; s > 0; s >>= 1) {
      if (tid < s) {
        float ov = rv[tid + s]; int oi = ri[tid + s];
        if (ov > rv[tid] || (ov == rv[tid] && oi < ri[tid])) { rv[tid] = ov; ri[tid] = oi; }
      }
      __syncthreads();
    }
    if (tid == 0) { topk[(size_t)i * TOPKK + k] = ri[0]; cosv[ri[0]] = -3.0e38f; }
    __syncthreads();
  }
}

// ===================== attention scores (per node, per head) ===============
__global__ void att_scores_kernel(
    const float* __restrict__ xh, const float* __restrict__ emb,
    const float* __restrict__ att_i, const float* __restrict__ att_j,
    const float* __restrict__ att_em_i, const float* __restrict__ att_em_j,
    float* __restrict__ si, float* __restrict__ sj, int BN, int N, int H, int D)
{
  int t = blockIdx.x * blockDim.x + threadIdx.x;
  if (t >= BN * H) return;
  const int bi = t / H, h = t - bi * H;
  const int n = bi % N;
  const float* xr = xh + (size_t)bi * H * D + (size_t)h * D;
  const float* er = emb + (size_t)n * D;
  float a = 0.f, b = 0.f;
  for (int d = 0; d < D; ++d) {
    const float xv = xr[d], ev = er[d];
    a += xv * att_i[h * D + d] + ev * att_em_i[h * D + d];
    b += xv * att_j[h * D + d] + ev * att_em_j[h * D + d];
  }
  si[t] = a; sj[t] = b;
}

// ===================== GAT softmax-aggregate ===============================
// 8 target nodes per block (one wave each). 21 edges/node (20 top-k + self);
// top-k self-edges masked to -1e30 (remove_self_loops semantics).
__global__ __launch_bounds__(256) void gat_agg_kernel(
    const float* __restrict__ xh, const float* __restrict__ si,
    const float* __restrict__ sj, const int* __restrict__ topk,
    const float* __restrict__ bias_gnn, float* __restrict__ agg,
    int B, int N, int H, int D)
{
  const int E = TOPKK + 1;             // 21
  __shared__ float s_sc[8][TOPKK + 1][4];
  __shared__ int   s_bj[8][TOPKK + 1];
  const int tid = threadIdx.x;
  const long base = (long)blockIdx.x * 8;
  // Stage A: per-edge, per-head raw scores
  if (tid < 8 * E) {
    const int nd = tid / E, e = tid - nd * E;
    const long bi = base + nd;
    const int  n  = (int)(bi % N);
    const long b  = bi / N;
    const int  j  = (e < TOPKK) ? topk[(size_t)n * TOPKK + e] : n;
    const bool dead = (e < TOPKK) && (j == n);
    const long bj = b * N + j;
    s_bj[nd][e] = (int)bj;
    #pragma unroll
    for (int h = 0; h < 4; ++h) {
      float a = si[bi * 4 + h] + sj[bj * 4 + h];
      a = (a > 0.f) ? a : 0.2f * a;                 // leaky_relu(0.2)
      s_sc[nd][e][h] = dead ? -1e30f : a;
    }
  }
  __syncthreads();
  // Stage B: per-(node, head) softmax over 21 edges
  if (tid < 8 * 4) {
    const int nd = tid >> 2, h = tid & 3;
    float m = -3.0e38f;
    for (int e = 0; e < E; ++e) m = fmaxf(m, s_sc[nd][e][h]);
    float s = 0.f;
    for (int e = 0; e < E; ++e) { float w = __expf(s_sc[nd][e][h] - m); s_sc[nd][e][h] = w; s += w; }
    const float inv = 1.f / (s + 1e-16f);
    for (int e = 0; e < E; ++e) s_sc[nd][e][h] *= inv;
  }
  __syncthreads();
  // Stage C: weighted gather-sum; lane owns 8 contiguous dims (one head)
  const int nd = tid >> 5, lane = tid & 31;
  const long bi = base + nd;
  const int h  = lane >> 3;            // D=64 -> 8 lanes per head
  const int d0 = lane * 8;
  float acc[8];
  #pragma unroll
  for (int u = 0; u < 8; ++u) acc[u] = 0.f;
  for (int e = 0; e < E; ++e) {
    const float w = s_sc[nd][e][h];
    const float* src = xh + (size_t)s_bj[nd][e] * (H * D) + d0;
    #pragma unroll
    for (int u = 0; u < 8; ++u) acc[u] += w * src[u];
  }
  float* dst = agg + (size_t)bi * (H * D) + d0;
  #pragma unroll
  for (int u = 0; u < 8; ++u) dst[u] = acc[u] + bias_gnn[d0 + u];
}

// ===================== BatchNorm stats (block per channel) =================
__global__ __launch_bounds__(256) void bn_stats_kernel(
    const float* __restrict__ x, float* __restrict__ mean,
    float* __restrict__ var, int rows, int C)
{
  __shared__ float ss[256], sq[256];
  const int c = blockIdx.x, tid = threadIdx.x;
  float s = 0.f, q = 0.f;
  for (int r = tid; r < rows; r += 256) {
    const float v = x[(size_t)r * C + c];
    s += v; q += v * v;
  }
  ss[tid] = s; sq[tid] = q;
  __syncthreads();
  for (int st = 128; st > 0; st >>= 1) {
    if (tid < st) { ss[tid] += ss[tid + st]; sq[tid] += sq[tid + st]; }
    __syncthreads();
  }
  if (tid == 0) {
    const float m = ss[0] / rows;
    mean[c] = m;
    var[c]  = sq[0] / rows - m * m;   // biased var (jnp.var default)
  }
}

// ===================== BN1 apply + ReLU + *emb =============================
__global__ void bn1_apply_kernel(
    const float* __restrict__ ff, const float* __restrict__ mean,
    const float* __restrict__ var, const float* __restrict__ g,
    const float* __restrict__ bta, const float* __restrict__ emb,
    float* __restrict__ h, long total, int N, int C)
{
  long t = (long)blockIdx.x * blockDim.x + threadIdx.x;
  if (t >= total) return;
  const long bi = t / C; const int c = (int)(t - bi * C);
  const int n = (int)(bi % N);
  float v = (ff[t] - mean[c]) * rsqrtf(var[c] + EPSB) * g[c] + bta[c];
  v = fmaxf(v, 0.f);
  h[t] = v * emb[(size_t)n * C + c];
}

// ===================== BN2 apply + ReLU ====================================
__global__ void bn2_apply_kernel(
    const float* __restrict__ hin, const float* __restrict__ mean,
    const float* __restrict__ var, const float* __restrict__ g,
    const float* __restrict__ bta, float* __restrict__ hout, long total, int C)
{
  long t = (long)blockIdx.x * blockDim.x + threadIdx.x;
  if (t >= total) return;
  const int c = (int)(t % C);
  const float v = (hin[t] - mean[c]) * rsqrtf(var[c] + EPSB) * g[c] + bta[c];
  hout[t] = fmaxf(v, 0.f);
}

// ===========================================================================
extern "C" void kernel_launch(void* const* d_in, const int* in_sizes, int n_in,
                              void* d_out, int out_size, void* d_ws, size_t ws_size,
                              hipStream_t stream)
{
  (void)in_sizes; (void)n_in; (void)out_size; (void)ws_size;
  constexpr int B = 16, N = 2000, F = 128, D = 64, H = 4, OW = 200;
  constexpr int BN = B * N;              // 32000
  constexpr int HD = H * D;              // 256

  const float* data     = (const float*)d_in[0];   // (B,N,F)
  const float* emb      = (const float*)d_in[1];   // (N,D)
  const float* W_lin    = (const float*)d_in[2];   // (F,HD)
  const float* att_i    = (const float*)d_in[3];   // (H,D)
  const float* att_j    = (const float*)d_in[4];
  const float* att_em_i = (const float*)d_in[5];
  const float* att_em_j = (const float*)d_in[6];
  const float* bias_gnn = (const float*)d_in[7];   // (HD)
  const float* W_ff     = (const float*)d_in[8];   // (HD,D)
  const float* b_ff     = (const float*)d_in[9];   // (D)
  const float* bn1_g    = (const float*)d_in[10];
  const float* bn1_b    = (const float*)d_in[11];
  const float* bn2_g    = (const float*)d_in[12];
  const float* bn2_b    = (const float*)d_in[13];
  const float* W_out    = (const float*)d_in[14];  // (D,OW)
  const float* b_out    = (const float*)d_in[15];  // (OW)
  float* out = (float*)d_out;                      // (B,N,OW)

  // ---- workspace layout (bytes) ----
  char* ws = (char*)d_ws;
  const size_t OFF_XH  = 0;                              // 32,768,000
  const size_t OFF_AGG = 34ull * 1024 * 1024;            // 32,768,000
  const size_t OFF_FF  = 68ull * 1024 * 1024;            // 8,192,000
  const size_t OFF_H1  = OFF_FF + 9ull * 1024 * 1024;    // 8,192,000
  const size_t OFF_H2  = OFF_H1 + 9ull * 1024 * 1024;    // 8,192,000
  const size_t OFF_SI  = OFF_H2 + 9ull * 1024 * 1024;    // BN*H f32 = 512,000
  const size_t OFF_SJ  = OFF_SI + 1024 * 1024;
  const size_t OFF_NRM = OFF_SJ + 1024 * 1024;           // N f32
  const size_t OFF_TK  = OFF_NRM + 64 * 1024;            // N*20 int = 160,000
  const size_t OFF_M1  = OFF_TK + 256 * 1024;            // 4 x 64 f32 stats
  float* xh   = (float*)(ws + OFF_XH);
  float* agg  = (float*)(ws + OFF_AGG);
  float* ff   = (float*)(ws + OFF_FF);
  float* h1   = (float*)(ws + OFF_H1);
  float* h2   = (float*)(ws + OFF_H2);
  float* si   = (float*)(ws + OFF_SI);
  float* sj   = (float*)(ws + OFF_SJ);
  float* nrm  = (float*)(ws + OFF_NRM);
  int*   topk = (int*)  (ws + OFF_TK);
  float* m1   = (float*)(ws + OFF_M1);
  float* v1   = m1 + 64;
  float* m2   = v1 + 64;
  float* v2   = m2 + 64;

  // 1) emb row norms
  row_norm_kernel<<<(N + 255) / 256, 256, 0, stream>>>(emb, nrm, N, D);
  // 2) cosine top-20 per row
  topk_cos_kernel<<<N, 256, 0, stream>>>(emb, nrm, topk, N, D);
  // 3) xh = data @ W_lin           (M=32000, N=256, K=128)
  wmma_gemm_f16<<<dim3((HD + 63) / 64, BN / 16), 128, 0, stream>>>(
      data, W_lin, nullptr, xh, BN, HD, F);
  // 4) per-(node,head) attention scores
  att_scores_kernel<<<(BN * H + 255) / 256, 256, 0, stream>>>(
      xh, emb, att_i, att_j, att_em_i, att_em_j, si, sj, BN, N, H, D);
  // 5) GAT softmax-aggregate (+bias_gnn)
  gat_agg_kernel<<<BN / 8, 256, 0, stream>>>(
      xh, si, sj, topk, bias_gnn, agg, B, N, H, D);
  // 6) ff = agg @ W_ff + b_ff      (M=32000, N=64, K=256)
  wmma_gemm_f16<<<dim3((D + 63) / 64, BN / 16), 128, 0, stream>>>(
      agg, W_ff, b_ff, ff, BN, D, HD);
  // 7) BN1 stats
  bn_stats_kernel<<<D, 256, 0, stream>>>(ff, m1, v1, BN, D);
  // 8) BN1 apply + relu + *emb
  bn1_apply_kernel<<<(BN * D + 255) / 256, 256, 0, stream>>>(
      ff, m1, v1, bn1_g, bn1_b, emb, h1, (long)BN * D, N, D);
  // 9) BN2 stats
  bn_stats_kernel<<<D, 256, 0, stream>>>(h1, m2, v2, BN, D);
  // 10) BN2 apply + relu
  bn2_apply_kernel<<<(BN * D + 255) / 256, 256, 0, stream>>>(
      h1, m2, v2, bn2_g, bn2_b, h2, (long)BN * D, D);
  // 11) out = h2 @ W_out + b_out   (M=32000, N=200, K=64)
  wmma_gemm_f16<<<dim3((OW + 63) / 64, BN / 16), 128, 0, stream>>>(
      h2, W_out, b_out, out, BN, OW, D);
}
